// Transformer_53463752901443
// MI455X (gfx1250) — matrix operand order repaired and checked by reading, not compiled, not run on hardware
//
#include <hip/hip_runtime.h>
#include <hip/hip_bf16.h>

typedef __attribute__((ext_vector_type(16))) _Float16 v16h;
typedef __attribute__((ext_vector_type(8)))  float    v8f;

#define LL long long

namespace {
constexpr int NL = 6, D = 512, H = 8, DH = 64, DFF = 2048, T = 512, S = 2048, NB = 4;
constexpr int TB = T * NB;  // 2048 rows of activations
}

// ---------------------------------------------------------------------------
// Batched strided GEMM on the WMMA unit.
// C[z][m,n] = act( alpha * sum_k A[z][m,k]*B[z][k,n] + bias ),  C row stride Cs_r,
// A rows K-contiguous (always true in this network). BVEC: B K-contiguous.
// Per-z base offsets: off = (z/zdiv)*b1 + (z%zdiv)*b2  (handles (b,h) batching).
// One wave computes a 16x64 tile: one A fragment reused across 4 B fragments
// (4 v_wmma per K-step); b128 loads; rolling pointers (no per-iter muls).
// ---------------------------------------------------------------------------
template <bool BVEC>
__global__ void __launch_bounds__(128)
wmma_gemm_kernel(const float* __restrict__ A, const float* __restrict__ Bm,
                 const float* __restrict__ bias, float* __restrict__ C,
                 int M, int K,
                 LL As_r, LL Ab1, LL Ab2,
                 LL Bs_r, LL Bs_c, LL Bb1, LL Bb2,
                 LL Cs_r, LL Cb1, LL Cb2,
                 int zdiv, float alpha, int bias_mode, int relu)
{
  const int z = blockIdx.z;
  const LL Aoff = (LL)(z / zdiv) * Ab1 + (LL)(z % zdiv) * Ab2;
  const LL Boff = (LL)(z / zdiv) * Bb1 + (LL)(z % zdiv) * Bb2;
  const LL Coff = (LL)(z / zdiv) * Cb1 + (LL)(z % zdiv) * Cb2;
  const int tm = blockIdx.y * blockDim.y + threadIdx.y;   // uniform per wave
  if (tm * 16 >= M) return;
  const int tn4  = blockIdx.x;      // 64-wide N tile
  const int lane = threadIdx.x;
  const int half = lane >> 4;       // 0: lanes 0-15, 1: lanes 16-31
  const int l    = lane & 15;

  // A fragment lane pointers: K runs [half*16, +8) and [half*16+8, +8).
  const float* a0p = A + Aoff + (LL)(tm * 16 + l) * As_r + half * 16;
  const float* a1p = a0p + 8;
  // B fragment lane pointers: 16 consecutive K starting at half*16, column = l.
  const float* bp[4];
#pragma unroll
  for (int t = 0; t < 4; ++t)
    bp[t] = Bm + Boff + (LL)(tn4 * 64 + t * 16 + l) * Bs_c +
            (BVEC ? (LL)(half * 16) : (LL)(half * 16) * Bs_r);
  const LL bstep = BVEC ? 32 : 32 * Bs_r;

  v8f acc[4] = {v8f{}, v8f{}, v8f{}, v8f{}};
  const int nk = K >> 5;
  for (int it = 0; it < nk; ++it) {
    if (it + 1 < nk) {   // speculative prefetch of next K slab (global_prefetch)
      __builtin_prefetch(a0p + 32, 0, 1);
      __builtin_prefetch(bp[0] + bstep, 0, 1);
    }
    v16h af;
    {
      const float4* r0 = (const float4*)a0p;
      const float4* r1 = (const float4*)a1p;
      float4 a0 = r0[0], a1 = r0[1], a2 = r1[0], a3 = r1[1];
      af[0]  = (_Float16)a0.x; af[1]  = (_Float16)a0.y;
      af[2]  = (_Float16)a0.z; af[3]  = (_Float16)a0.w;
      af[4]  = (_Float16)a1.x; af[5]  = (_Float16)a1.y;
      af[6]  = (_Float16)a1.z; af[7]  = (_Float16)a1.w;
      af[8]  = (_Float16)a2.x; af[9]  = (_Float16)a2.y;
      af[10] = (_Float16)a2.z; af[11] = (_Float16)a2.w;
      af[12] = (_Float16)a3.x; af[13] = (_Float16)a3.y;
      af[14] = (_Float16)a3.z; af[15] = (_Float16)a3.w;
    }
#pragma unroll
    for (int t = 0; t < 4; ++t) {
      v16h bf;
      if (BVEC) {
        const float4* br = (const float4*)bp[t];
        float4 b0 = br[0], b1 = br[1], b2 = br[2], b3 = br[3];
        bf[0]  = (_Float16)b0.x; bf[1]  = (_Float16)b0.y;
        bf[2]  = (_Float16)b0.z; bf[3]  = (_Float16)b0.w;
        bf[4]  = (_Float16)b1.x; bf[5]  = (_Float16)b1.y;
        bf[6]  = (_Float16)b1.z; bf[7]  = (_Float16)b1.w;
        bf[8]  = (_Float16)b2.x; bf[9]  = (_Float16)b2.y;
        bf[10] = (_Float16)b2.z; bf[11] = (_Float16)b2.w;
        bf[12] = (_Float16)b3.x; bf[13] = (_Float16)b3.y;
        bf[14] = (_Float16)b3.z; bf[15] = (_Float16)b3.w;
      } else {
        const float* q = bp[t];
#pragma unroll
        for (int u = 0; u < 16; ++u) bf[u] = (_Float16)q[(LL)u * Bs_r];
      }
      acc[t] = __builtin_amdgcn_wmma_f32_16x16x32_f16(false, af, false, bf, (short)0,
                                                      acc[t], false, false);
      bp[t] += bstep;
    }
    a0p += 32; a1p += 32;
  }
#pragma unroll
  for (int t = 0; t < 4; ++t) {
    int col = tn4 * 64 + t * 16 + l;
#pragma unroll
    for (int r = 0; r < 8; ++r) {
      int row = tm * 16 + half * 8 + r;        // C/D layout: vgpr r -> M = half*8 + r
      float v = acc[t][r] * alpha;
      if (bias_mode == 1)      v += bias[col]; // per-output-feature (N)
      else if (bias_mode == 2) v += bias[row]; // per-channel (M), conv1d bias
      if (relu) v = fmaxf(v, 0.f);
      C[Coff + (LL)row * Cs_r + col] = v;
    }
  }
}

// ---------------------------------------------------------------------------
// wave32 reductions
// ---------------------------------------------------------------------------
__device__ inline float wred_sum(float v) {
#pragma unroll
  for (int o = 16; o; o >>= 1) v += __shfl_xor(v, o, 32);
  return v;
}
__device__ inline float wred_max(float v) {
#pragma unroll
  for (int o = 16; o; o >>= 1) v = fmaxf(v, __shfl_xor(v, o, 32));
  return v;
}

// Streaming in-place row softmax (wave per row), arbitrary row length
__global__ void softmax_rows_kernel(float* __restrict__ x, int rows, int n) {
  int row = blockIdx.x * blockDim.y + threadIdx.y;
  if (row >= rows) return;
  float* xr = x + (LL)row * n;
  int lane = threadIdx.x;
  float mx = -3.0e38f;
  for (int j = lane; j < n; j += 32) mx = fmaxf(mx, xr[j]);
  mx = wred_max(mx);
  float s = 0.f;
  for (int j = lane; j < n; j += 32) s += __expf(xr[j] - mx);
  s = wred_sum(s);
  float inv = 1.0f / s;
  for (int j = lane; j < n; j += 32) xr[j] = __expf(xr[j] - mx) * inv;
}

// y[row,:] = LN(x[row,:] + residual, g, b)  over D=512; wave per row.
// res_mode: 0 none, 1 full [rows,D], 2 broadcast r[(row%NB)*D + d]
__global__ void ln_add_kernel(const float* __restrict__ x, const float* __restrict__ r,
                              const float* __restrict__ g, const float* __restrict__ be,
                              float* __restrict__ y, int rows, int res_mode) {
  int row = blockIdx.x * blockDim.y + threadIdx.y;
  if (row >= rows) return;
  int lane = threadIdx.x;
  const float* xr = x + (LL)row * D;
  const float* rr = nullptr;
  if (res_mode == 1)      rr = r + (LL)row * D;
  else if (res_mode == 2) rr = r + (LL)(row % NB) * D;
  float v[16];
  float s = 0.f;
#pragma unroll
  for (int i = 0; i < 16; ++i) {
    int idx = lane + 32 * i;
    float t = xr[idx];
    if (rr) t += rr[idx];
    v[i] = t; s += t;
  }
  s = wred_sum(s);
  float mean = s * (1.f / D);
  float vs = 0.f;
#pragma unroll
  for (int i = 0; i < 16; ++i) { float d2 = v[i] - mean; vs += d2 * d2; }
  vs = wred_sum(vs);
  float rstd = rsqrtf(vs * (1.f / D) + 1e-5f);
#pragma unroll
  for (int i = 0; i < 16; ++i) {
    int idx = lane + 32 * i;
    y[(LL)row * D + idx] = (v[i] - mean) * rstd * g[idx] + be[idx];
  }
}

// xsum[b*D+d] = sum_t x[t,b,d]   (x flat [T,NB,D] => stride T is NB*D)
__global__ void xsum_kernel(const float* __restrict__ x, float* __restrict__ xs) {
  int idx = blockIdx.x * blockDim.x + threadIdx.x;   // NB*D = 2048
  if (idx >= NB * D) return;
  float s = 0.f;
  for (int t = 0; t < T; ++t) s += x[(LL)t * (NB * D) + idx];
  xs[idx] = s;
}

// qs[b,o] = sum_k Wq[o,k]*xsum[b,k] + T*bq[o]   (wave per output)
__global__ void qs_kernel(const float* __restrict__ w0, const float* __restrict__ cb0,
                          const float* __restrict__ xs, float* __restrict__ qs) {
  int row = blockIdx.x * blockDim.y + threadIdx.y;   // NB*D rows
  if (row >= NB * D) return;
  int b = row >> 9, o = row & (D - 1);
  int lane = threadIdx.x;
  float acc = 0.f;
  for (int k = lane; k < D; k += 32) acc += w0[(LL)o * D + k] * xs[b * D + k];
  acc = wred_sum(acc);
  if (lane == 0) qs[row] = acc + (float)T * cb0[o];
}

// c[b,h,m] = (1/8) * sum_d qs[b, d*H+h] * k[b, d*H+h, m]
__global__ void cross_c_kernel(const float* __restrict__ qs, const float* __restrict__ k,
                               float* __restrict__ c) {
  int idx = blockIdx.x * blockDim.x + threadIdx.x;   // NB*H*S = 65536
  if (idx >= NB * H * S) return;
  int m = idx & (S - 1);
  int bh = idx >> 11;
  int b = bh >> 3, h = bh & 7;
  float acc = 0.f;
  for (int d = 0; d < DH; ++d) {
    int ch = d * H + h;
    acc += qs[b * D + ch] * k[((LL)(b * D + ch)) * S + m];
  }
  c[idx] = acc * 0.125f;   // 1/sqrt(DH)
}

// p[b,ch,:] = softmax_m( k[b,ch,m] * c[b, ch%H, m] )   (wave per row, streaming)
__global__ void cross_softmax_kernel(const float* __restrict__ k, const float* __restrict__ c,
                                     float* __restrict__ p) {
  int row = blockIdx.x * blockDim.y + threadIdx.y;   // NB*D = 2048 rows
  if (row >= NB * D) return;
  int b = row >> 9, ch = row & (D - 1), h = ch & 7;
  int lane = threadIdx.x;
  const float* kr = k + (LL)row * S;
  const float* cr = c + (LL)(b * H + h) * S;
  float mx = -3.0e38f;
  for (int j = lane; j < S; j += 32) mx = fmaxf(mx, kr[j] * cr[j]);
  mx = wred_max(mx);
  float s = 0.f;
  for (int j = lane; j < S; j += 32) s += __expf(kr[j] * cr[j] - mx);
  s = wred_sum(s);
  float inv = 1.0f / s;
  for (int j = lane; j < S; j += 32) p[(LL)row * S + j] = __expf(kr[j] * cr[j] - mx) * inv;
}

// pw[b,h,m] = sum_n dw[n] * p[b, n*H+h, m]
__global__ void pw_kernel(const float* __restrict__ dw, const float* __restrict__ p,
                          float* __restrict__ pw) {
  int idx = blockIdx.x * blockDim.x + threadIdx.x;   // NB*H*S
  if (idx >= NB * H * S) return;
  int m = idx & (S - 1);
  int bh = idx >> 11;
  int b = bh >> 3, h = bh & 7;
  float acc = 0.f;
  for (int n = 0; n < DH; ++n)
    acc += dw[n] * p[((LL)(b * D + n * H + h)) * S + m];
  pw[idx] = acc;
}

// cross[b,ch] = db + sum_m v[b,ch,m] * pw[b, ch%H, m]   (wave per row)
__global__ void oo_kernel(const float* __restrict__ v, const float* __restrict__ pw,
                          const float* __restrict__ db, float* __restrict__ cr) {
  int row = blockIdx.x * blockDim.y + threadIdx.y;   // NB*D rows
  if (row >= NB * D) return;
  int b = row >> 9, ch = row & (D - 1), h = ch & 7;
  int lane = threadIdx.x;
  const float* vr = v + (LL)row * S;
  const float* pr = pw + (LL)(b * H + h) * S;
  float acc = 0.f;
  for (int m = lane; m < S; m += 32) acc += vr[m] * pr[m];
  acc = wred_sum(acc);
  if (lane == 0) cr[row] = acc + db[0];
}

// ---------------------------------------------------------------------------
// Host side
// ---------------------------------------------------------------------------
static void gemm(hipStream_t st, const float* A, const float* Bm, const float* bias, float* C,
                 int M, int N, int K,
                 LL As_r, LL Ab1, LL Ab2,
                 LL Bs_r, LL Bs_c, LL Bb1, LL Bb2,
                 LL Cs_r, LL Cb1, LL Cb2,
                 int zdiv, int batch, float alpha, int bias_mode, int relu) {
  dim3 blk(32, 4, 1);
  dim3 grd(N / 64, (M / 16 + 3) / 4, batch);
  if (Bs_r == 1)
    wmma_gemm_kernel<true><<<grd, blk, 0, st>>>(A, Bm, bias, C, M, K,
        As_r, Ab1, Ab2, Bs_r, Bs_c, Bb1, Bb2, Cs_r, Cb1, Cb2,
        zdiv, alpha, bias_mode, relu);
  else
    wmma_gemm_kernel<false><<<grd, blk, 0, st>>>(A, Bm, bias, C, M, K,
        As_r, Ab1, Ab2, Bs_r, Bs_c, Bb1, Bb2, Cs_r, Cb1, Cb2,
        zdiv, alpha, bias_mode, relu);
}

extern "C" void kernel_launch(void* const* d_in, const int* in_sizes, int n_in,
                              void* d_out, int out_size, void* d_ws, size_t ws_size,
                              hipStream_t stream) {
  (void)in_sizes; (void)n_in; (void)out_size; (void)ws_size;
  const float* tgt  = (const float*)d_in[0];
  const float* mem  = (const float*)d_in[1];
  const float* sa_w = (const float*)d_in[2];
  const float* sa_b = (const float*)d_in[3];
  const float* sa_ow= (const float*)d_in[4];
  const float* sa_ob= (const float*)d_in[5];
  const float* cx_w = (const float*)d_in[6];
  const float* cx_b = (const float*)d_in[7];
  const float* dm_w = (const float*)d_in[8];
  const float* dm_b = (const float*)d_in[9];
  const float* l1_w = (const float*)d_in[10];
  const float* l1_b = (const float*)d_in[11];
  const float* l2_w = (const float*)d_in[12];
  const float* l2_b = (const float*)d_in[13];
  const float* n1_g = (const float*)d_in[14];
  const float* n1_b = (const float*)d_in[15];
  const float* n2_g = (const float*)d_in[16];
  const float* n2_b = (const float*)d_in[17];
  const float* n3_g = (const float*)d_in[18];
  const float* n3_b = (const float*)d_in[19];
  const float* nf_g = (const float*)d_in[20];
  const float* nf_b = (const float*)d_in[21];
  float* out = (float*)d_out;

  float* ws = (float*)d_ws;
  size_t off = 0;
  float* x      = ws + off; off += (size_t)TB * D;          // activations [T*NB, D]
  float* tmp    = ws + off; off += (size_t)TB * D;          // sublayer output
  float* qkv    = ws + off; off += (size_t)TB * 3 * D;      // [T*NB, 3D]
  float* scores = ws + off; off += (size_t)NB * H * T * T;  // also pbuf / ffn hidden
  float* attn   = ws + off; off += (size_t)TB * D;
  float* kbuf   = ws + off; off += (size_t)NB * D * S;
  float* vbuf   = ws + off; off += (size_t)NB * D * S;
  float* cbuf   = ws + off; off += (size_t)NB * H * S;
  float* pwb    = ws + off; off += (size_t)NB * H * S;
  float* xs     = ws + off; off += (size_t)NB * D;
  float* qsb    = ws + off; off += (size_t)NB * D;
  float* crossv = ws + off; off += (size_t)NB * D;
  float* pbuf   = scores;   // cross p [NB, D, S]: lives where scores lived
  float* hidden = scores;   // FFN hidden [TB, DFF]: lives where scores lived

  hipMemcpyAsync(x, tgt, (size_t)TB * D * sizeof(float), hipMemcpyDeviceToDevice, stream);

  dim3 wblk(32, 8, 1);   // 8 waves per block for wave-per-row kernels

  for (int i = 0; i < NL; ++i) {
    const float* wqkv = sa_w + (LL)i * 3 * D * D;
    const float* bqkv = sa_b + (LL)i * 3 * D;
    const float* wo   = sa_ow + (LL)i * D * D;
    const float* bo   = sa_ob + (LL)i * D;
    const float* cw   = cx_w + (LL)i * 3 * D * D;
    const float* cb   = cx_b + (LL)i * 3 * D;
    const float* dw   = dm_w + (LL)i * DH;
    const float* dbp  = dm_b + i;

    // ---- self-attention ----
    // qkv = x @ sa_w^T + sa_b : [2048,1536]
    gemm(stream, x, wqkv, bqkv, qkv, TB, 3 * D, D,
         D, 0, 0,  1, D, 0, 0,  3 * D, 0, 0,  1, 1, 1.f, 1, 0);
    // scores[z=(b,h)] = Q @ K^T / 8 : 32 x [512,512], K dim 64
    gemm(stream, qkv, qkv + D, nullptr, scores, T, T, DH,
         (LL)NB * 3 * D, 3 * D, DH,
         1, (LL)NB * 3 * D, 3 * D, DH,
         T, (LL)H * T * T, (LL)T * T,
         H, NB * H, 0.125f, 0, 0);
    softmax_rows_kernel<<<dim3(NB * H * T / 8), wblk, 0, stream>>>(scores, NB * H * T, T);
    // attn[z] = P @ V : 32 x [512,64]  (strided-B instantiation)
    gemm(stream, scores, qkv + 2 * D, nullptr, attn, T, DH, T,
         T, (LL)H * T * T, (LL)T * T,
         (LL)NB * 3 * D, 1, 3 * D, DH,
         (LL)NB * D, D, DH,
         H, NB * H, 1.f, 0, 0);
    // out proj: tmp = attn @ ow^T + ob
    gemm(stream, attn, wo, bo, tmp, TB, D, D,
         D, 0, 0,  1, D, 0, 0,  D, 0, 0,  1, 1, 1.f, 1, 0);
    // x = LN(x + tmp)
    ln_add_kernel<<<dim3(TB / 8), wblk, 0, stream>>>(x, tmp, n1_g + (LL)i * D,
                                                     n1_b + (LL)i * D, x, TB, 1);

    // ---- custom cross-attention (algebraically collapsed) ----
    xsum_kernel<<<dim3(NB * D / 256), dim3(256), 0, stream>>>(x, xs);
    qs_kernel<<<dim3(NB * D / 8), wblk, 0, stream>>>(cw, cb, xs, qsb);
    // K,V conv1d projections: per batch z, W[D,D] @ mem_z[D,S] + bias(per-row)
    gemm(stream, cw + (LL)D * D, mem, cb + D, kbuf, D, S, D,
         D, 0, 0,
         1, (LL)NB * D, D, 0,
         S, (LL)D * S, 0,
         1, NB, 1.f, 2, 0);
    gemm(stream, cw + (LL)2 * D * D, mem, cb + 2 * D, vbuf, D, S, D,
         D, 0, 0,
         1, (LL)NB * D, D, 0,
         S, (LL)D * S, 0,
         1, NB, 1.f, 2, 0);
    cross_c_kernel<<<dim3(NB * H * S / 256), dim3(256), 0, stream>>>(qsb, kbuf, cbuf);
    cross_softmax_kernel<<<dim3(NB * D / 8), wblk, 0, stream>>>(kbuf, cbuf, pbuf);
    pw_kernel<<<dim3(NB * H * S / 256), dim3(256), 0, stream>>>(dw, pbuf, pwb);
    oo_kernel<<<dim3(NB * D / 8), wblk, 0, stream>>>(vbuf, pwb, dbp, crossv);
    // x = LN(x + broadcast(crossv))
    ln_add_kernel<<<dim3(TB / 8), wblk, 0, stream>>>(x, crossv, n2_g + (LL)i * D,
                                                     n2_b + (LL)i * D, x, TB, 2);

    // ---- FFN ----
    gemm(stream, x, l1_w + (LL)i * DFF * D, l1_b + (LL)i * DFF, hidden, TB, DFF, D,
         D, 0, 0,  1, D, 0, 0,  DFF, 0, 0,  1, 1, 1.f, 1, 1 /*relu*/);
    gemm(stream, hidden, l2_w + (LL)i * D * DFF, l2_b + (LL)i * D, tmp, TB, D, DFF,
         DFF, 0, 0,  1, DFF, 0, 0,  D, 0, 0,  1, 1, 1.f, 1, 0);
    ln_add_kernel<<<dim3(TB / 8), wblk, 0, stream>>>(x, tmp, n3_g + (LL)i * D,
                                                     n3_b + (LL)i * D, x, TB, 1);
  }

  // final LN -> out [1,T,NB,D]
  ln_add_kernel<<<dim3(TB / 8), wblk, 0, stream>>>(x, nullptr, nf_g, nf_b, out, TB, 0);
}
